// EGNNUnPooling_75359496176073
// MI455X (gfx1250) — compile-verified, matrix operand order checked by reading, not run on hardware
//
#include <hip/hip_runtime.h>
#include <math.h>

#define N_NODES   50000
#define N_EDGES   800000
#define HID       32
#define NUM_GRAPHS 16
#define N_LAYERS  4
#define N_TILES   (N_NODES / 16)   // 3125

typedef __attribute__((ext_vector_type(16))) _Float16 v16h;
typedef __attribute__((ext_vector_type(8)))  _Float16 v8h;
typedef __attribute__((ext_vector_type(8)))  float    v8f;
typedef __attribute__((ext_vector_type(4)))  float    v4f;
typedef __attribute__((ext_vector_type(4)))  unsigned int u32x4;
typedef __attribute__((ext_vector_type(8)))  int      i32x8;
typedef __attribute__((ext_vector_type(4)))  int      i32x4;

union F16Frag { v16h v; v8h h[2]; };

__device__ __forceinline__ float silu_f(float v){ return v * (1.0f / (1.0f + __expf(-v))); }

#define WMMA_F16(a,b,c) \
  __builtin_amdgcn_wmma_f32_16x16x32_f16(false,(a),false,(b),(short)0,(c),false,false)

// ---------------------------------------------------------------------------
// TDM: DMA a contiguous block (8-byte units) from global into LDS.
// D# per CDNA5 ISA §8: group0 = {count=1 | lds_addr | global_addr | type=2},
// group1 = {data_size=8B, tensor_dim0=n8, tensor_dim1=1, tile_dim0=n8,
//           tile_dim1=1, stride0=n8, stride1=n8}, groups 2/3 unused (2-D).
// LDS byte address = low 32 bits of the generic shared pointer (aperture rule).
// Issue from one wave; EXEC is ignored for tensor ops. Tracked by TENSORcnt.
// ---------------------------------------------------------------------------
__device__ __forceinline__ void tdm_load_to_lds(void* lds_dst, const void* gsrc,
                                                unsigned nbytes) {
  const unsigned n8 = nbytes >> 3;                 // 8-byte elements (<= 65535)
  u32x4 g0;
  g0[0] = 1u;                                      // count=1, user descriptor
  g0[1] = (unsigned)(unsigned long long)(uintptr_t)lds_dst;   // lds_addr
  unsigned long long ga = (unsigned long long)(uintptr_t)gsrc;
  g0[2] = (unsigned)ga;                            // global_addr[31:0]
  g0[3] = ((unsigned)(ga >> 32) & 0x01FFFFFFu) | (2u << 30);  // addr[56:32] | type=2
  i32x8 g1;
  g1[0] = 0x00030000;                              // data_size = 3 (8B)
  g1[1] = (int)((n8 & 0xFFFFu) << 16);             // tensor_dim0[15:0]
  g1[2] = (int)((n8 >> 16) | (1u << 16));          // tensor_dim0[31:16] | tensor_dim1 lo = 1
  g1[3] = (int)((n8 & 0xFFFFu) << 16);             // tensor_dim1 hi = 0 | tile_dim0 = n8
  g1[4] = 1;                                       // tile_dim1 = 1, tile_dim2 = 0
  g1[5] = (int)n8;                                 // tensor_dim0_stride lo
  g1[6] = (int)((n8 & 0xFFFFu) << 16);             // stride0 hi = 0 | stride1 lo
  g1[7] = (int)(n8 >> 16);                         // stride1 hi
  i32x4 z4 = {};
#if __clang_major__ >= 23
  i32x8 z8 = {};
  __builtin_amdgcn_tensor_load_to_lds(g0, g1, z4, z4, z8, 0);
#else
  __builtin_amdgcn_tensor_load_to_lds(g0, g1, z4, z4, 0);
#endif
  __builtin_amdgcn_s_wait_tensorcnt((short)0);
}

// ---------------------------------------------------------------------------
// WMMA fragment helpers (16x16x32 f16, wave32).
// A (16Mx32K): lane r=lane&15 -> M=r; half=lane>>4 picks K {0..7,16..23} or
// {8..15,24..31}. B (32Kx16N) loaded from an [N][K] transposed-weight tile
// with identical indexing (N=r). C/D v8f: VGPR i -> M=half*8+i, N=r.
// ---------------------------------------------------------------------------
__device__ __forceinline__ v16h frag_from_global_f32(const float* __restrict__ base,
                                                     int row, int half) {
  const float* p = base + (size_t)row * HID + half * 8;
  v4f a0 = *(const v4f*)(p);      v4f a1 = *(const v4f*)(p + 4);
  v4f b0 = *(const v4f*)(p + 16); v4f b1 = *(const v4f*)(p + 20);
  F16Frag f;
#pragma unroll
  for (int i = 0; i < 4; ++i) {
    f.v[i]      = (_Float16)a0[i];  f.v[4 + i]  = (_Float16)a1[i];
    f.v[8 + i]  = (_Float16)b0[i];  f.v[12 + i] = (_Float16)b1[i];
  }
  return f.v;
}

__device__ __forceinline__ v16h frag_from_global_f16(const _Float16* __restrict__ base,
                                                     int row, int half) {
  const _Float16* p = base + (size_t)row * HID + half * 8;
  F16Frag f;
  f.h[0] = *(const v8h*)(p);
  f.h[1] = *(const v8h*)(p + 16);
  return f.v;
}

__device__ __forceinline__ v16h frag_from_lds(const _Float16* tile, int r, int half) {
  const _Float16* p = tile + r * HID + half * 8;
  F16Frag f;
  f.h[0] = *(const v8h*)(p);
  f.h[1] = *(const v8h*)(p + 16);
  return f.v;
}

__device__ __forceinline__ void stage_c(_Float16* tile /*[16M][32N]*/,
                                        const v8f& c, int r, int half, int ncolbase) {
  const int col = ncolbase + r;
#pragma unroll
  for (int i = 0; i < 8; ++i) tile[(half * 8 + i) * HID + col] = (_Float16)c[i];
}

// ---------------------------------------------------------------------------
// Weight prep: f16 transposed [N][K] blocks (1024 f16 each):
// per layer: 0-2 edge_w1t (h_row,h_col,eattr), 3 edge_w2t, 4 coord_w1t,
// 5 cross_w1t, 6 cross_w2t, 7-8 node_w1t, 9 node_w2t; then emb_in_wt,
// emb_out_wt. f32 vectors (32 each): 0 w1_radial, 1 edge_b1, 2 edge_b2,
// 3 coord_b1, 4 coord_w2, 5 cross_b1, 6 cross_b2, 7 cross_w3, 8 node_b1,
// 9 node_b2; then emb_in_b, emb_out_b.
// ---------------------------------------------------------------------------
struct PtrTab { const float* p[4 + N_LAYERS * 16]; };
// p[0]=emb_in_b p[1]=emb_in_w p[2]=emb_out_b p[3]=emb_out_w
// p[4+l*16+k], sorted keys: 0 coord_b1 1 coord_w1 2 coord_w2 3 cross_b1
// 4 cross_b2 5 cross_w1 6 cross_w2 7 cross_w3 8 edge_b1 9 edge_b2 10 edge_w1
// 11 edge_w2 12 node_b1 13 node_b2 14 node_w1 15 node_w2

__global__ void prep_weights_kernel(PtrTab tab, _Float16* __restrict__ wf16,
                                    float* __restrict__ wf32) {
  const int pk[10]   = {10, 10, 10, 11, 1, 5, 6, 14, 14, 15};
  const int roff[10] = { 0, 32, 65,  0, 0, 0, 0,  0, 32,  0};
  const int vk[10]   = {10,  8,  9,  0, 2, 3, 4,  7, 12, 13};
  int tid = blockIdx.x * blockDim.x + threadIdx.x;
  int stride = gridDim.x * blockDim.x;
  int total16 = (N_LAYERS * 10 + 2) * 1024;
  for (int idx = tid; idx < total16; idx += stride) {
    int blk = idx >> 10, e = idx & 1023;
    int n = e >> 5, k = e & 31;
    const float* src; int r0 = 0;
    if (blk < N_LAYERS * 10) {
      int l = blk / 10, b = blk % 10;
      src = tab.p[4 + l * 16 + pk[b]]; r0 = roff[b];
    } else {
      src = (blk == N_LAYERS * 10) ? tab.p[1] : tab.p[3];
    }
    wf16[idx] = (_Float16)src[(r0 + k) * HID + n];   // transpose: [N][K]
  }
  int total32 = (N_LAYERS * 10 + 2) * 32;
  for (int idx = tid; idx < total32; idx += stride) {
    int blk = idx >> 5, n = idx & 31;
    float v;
    if (blk < N_LAYERS * 10) {
      int l = blk / 10, b = blk % 10;
      const float* src = tab.p[4 + l * 16 + vk[b]];
      v = (b == 0) ? src[64 * HID + n] : src[n];     // b==0: edge_w1 radial row
    } else {
      v = (blk == N_LAYERS * 10) ? tab.p[0][n] : tab.p[2][n];
    }
    wf32[idx] = v;
  }
}

// ---------------------------------------------------------------------------
// One-time helpers
// ---------------------------------------------------------------------------
__global__ void f32_to_f16_kernel(const float* __restrict__ in,
                                  _Float16* __restrict__ out, int n) {
  int i = blockIdx.x * blockDim.x + threadIdx.x;
  int stride = gridDim.x * blockDim.x;
  for (; i < n; i += stride) out[i] = (_Float16)in[i];
}

__global__ void degree_kernel(const int* __restrict__ edges, float* __restrict__ tcnt) {
  int e = blockIdx.x * blockDim.x + threadIdx.x;
  if (e >= N_EDGES) return;
  atomicAdd(&tcnt[edges[e]], 1.0f);               // in-degree over `row`
}

__global__ void gcount_kernel(const int* __restrict__ batch, float* __restrict__ cnt) {
  int n = blockIdx.x * blockDim.x + threadIdx.x;
  if (n >= N_NODES) return;
  atomicAdd(&cnt[batch[n]], 1.0f);
}

// ---------------------------------------------------------------------------
// Per-layer graph coordinate means
// ---------------------------------------------------------------------------
__global__ void mean_accum_kernel(const float* __restrict__ x, const int* __restrict__ batch,
                                  float* __restrict__ csum) {
  int n = blockIdx.x * blockDim.x + threadIdx.x;
  if (n >= N_NODES) return;
  int g = batch[n];
  atomicAdd(&csum[g * 3 + 0], x[n * 3 + 0]);
  atomicAdd(&csum[g * 3 + 1], x[n * 3 + 1]);
  atomicAdd(&csum[g * 3 + 2], x[n * 3 + 2]);
}

__global__ void mean_div_kernel(const float* __restrict__ csum, const float* __restrict__ cnt,
                                float* __restrict__ mean) {
  int i = threadIdx.x;
  if (i >= NUM_GRAPHS * 3) return;
  float c = cnt[i / 3];
  mean[i] = csum[i] / ((c == 0.0f) ? 1.0f : c);
}

__global__ void coord_update_kernel(float* __restrict__ x, const float* __restrict__ tsum,
                                    const float* __restrict__ tcnt) {
  int n = blockIdx.x * blockDim.x + threadIdx.x;
  if (n >= N_NODES) return;
  float c = fmaxf(tcnt[n], 1.0f);
  x[n * 3 + 0] += tsum[n * 3 + 0] / c;
  x[n * 3 + 1] += tsum[n * 3 + 1] / c;
  x[n * 3 + 2] += tsum[n * 3 + 2] / c;
}

// ---------------------------------------------------------------------------
// Edge kernel: 16 edges per wave tile; 14 WMMAs per tile; f16 gathers;
// weights staged into LDS by the Tensor Data Mover.
// ---------------------------------------------------------------------------
__global__ void __launch_bounds__(256, 2)
egnn_edge_kernel(const _Float16* __restrict__ h_half, const float* __restrict__ x_cur,
                 const int* __restrict__ edges, const _Float16* __restrict__ eattr_half,
                 const int* __restrict__ batch, const float* __restrict__ mean,
                 const _Float16* __restrict__ wl16, const float* __restrict__ wl32,
                 float* __restrict__ tsum, float* __restrict__ magg) {
  __shared__ __align__(16) _Float16 sw[7 * 1024];       // w1t x3, w2t, cw1t, xw1t, xw2t
  __shared__ float svec[320];
  __shared__ __align__(16) _Float16 s_ef[8][16 * HID];  // edge_feat stage (per wave)
  __shared__ __align__(16) _Float16 s_tmp[8][16 * HID]; // scratch stage (per wave)
  __shared__ float s_rad[8][16];
  __shared__ int   s_row[8][16];

  const int tid = threadIdx.x;
  if (tid == 0) tdm_load_to_lds(sw, wl16, 7 * 1024 * 2);       // TENSOR_LOAD_TO_LDS
  for (int i = tid; i < 320; i += 256) svec[i] = wl32[i];
  __syncthreads();

  const int wave = tid >> 5, lane = tid & 31;
  const int r = lane & 15, half = lane >> 4;
  const int e0 = (blockIdx.x * 8 + wave) * 16;          // E % 128 == 0: no tail
  const int myE = e0 + r;
  const int myRow = edges[myE];
  const int myCol = edges[N_EDGES + myE];

  // per-edge geometry on lanes 0..15
  float dx = 0, dy = 0, dz = 0, cx = 0, cy = 0, cz = 0;
  if (half == 0) {
    const float* xr = x_cur + (size_t)myRow * 3;
    const float* xc = x_cur + (size_t)myCol * 3;
    float xr0 = xr[0], xr1 = xr[1], xr2 = xr[2];
    float xc0 = xc[0], xc1 = xc[1], xc2 = xc[2];
    dx = xr0 - xc0; dy = xr1 - xc1; dz = xr2 - xc2;
    s_rad[wave][r] = dx * dx + dy * dy + dz * dz;
    s_row[wave][r] = myRow;
    const float* mr = mean + 3 * batch[myRow];
    const float* mc = mean + 3 * batch[myCol];
    float ax = xr0 - mr[0], ay = xr1 - mr[1], az = xr2 - mr[2];
    float bx = xc0 - mc[0], by = xc1 - mc[1], bz = xc2 - mc[2];
    cx = ay * bz - az * by; cy = az * bx - ax * bz; cz = ax * by - ay * bx;
    float nrm = sqrtf(cx * cx + cy * cy + cz * cz) + 1.0f;
    cx /= nrm; cy /= nrm; cz /= nrm;
  }

  // edge MLP L1: concat(h[row],h[col],radial,edge_attr) @ W1  (radial = rank-1)
  v16h aHr = frag_from_global_f16(h_half, myRow, half);
  v16h aHc = frag_from_global_f16(h_half, myCol, half);
  v16h aEA = frag_from_global_f16(eattr_half, myE, half);
  v8f c_lo = {}, c_hi = {};
  c_lo = WMMA_F16(aHr, frag_from_lds(sw + 0,          r, half), c_lo);
  c_hi = WMMA_F16(aHr, frag_from_lds(sw + 512,        r, half), c_hi);
  c_lo = WMMA_F16(aHc, frag_from_lds(sw + 1024,       r, half), c_lo);
  c_hi = WMMA_F16(aHc, frag_from_lds(sw + 1024 + 512, r, half), c_hi);
  c_lo = WMMA_F16(aEA, frag_from_lds(sw + 2048,       r, half), c_lo);
  c_hi = WMMA_F16(aEA, frag_from_lds(sw + 2048 + 512, r, half), c_hi);
  {
    float wr_lo = svec[0 * 32 + r], wr_hi = svec[0 * 32 + 16 + r];
    float b_lo  = svec[1 * 32 + r], b_hi  = svec[1 * 32 + 16 + r];
#pragma unroll
    for (int i = 0; i < 8; ++i) {
      float rad = s_rad[wave][half * 8 + i];
      c_lo[i] = silu_f(c_lo[i] + rad * wr_lo + b_lo);
      c_hi[i] = silu_f(c_hi[i] + rad * wr_hi + b_hi);
    }
  }
  stage_c(s_tmp[wave], c_lo, r, half, 0);
  stage_c(s_tmp[wave], c_hi, r, half, 16);
  v16h aT = frag_from_lds(s_tmp[wave], r, half);

  // edge MLP L2 -> edge_feat
  v8f ef_lo = {}, ef_hi = {};
  ef_lo = WMMA_F16(aT, frag_from_lds(sw + 3 * 1024,       r, half), ef_lo);
  ef_hi = WMMA_F16(aT, frag_from_lds(sw + 3 * 1024 + 512, r, half), ef_hi);
  {
    float b_lo = svec[2 * 32 + r], b_hi = svec[2 * 32 + 16 + r];
#pragma unroll
    for (int i = 0; i < 8; ++i) { ef_lo[i] = silu_f(ef_lo[i] + b_lo); ef_hi[i] = silu_f(ef_hi[i] + b_hi); }
  }
  stage_c(s_ef[wave], ef_lo, r, half, 0);
  stage_c(s_ef[wave], ef_hi, r, half, 16);
  v16h aEF = frag_from_lds(s_ef[wave], r, half);

  // message aggregation: magg[row] += edge_feat
#pragma unroll
  for (int i = 0; i < 8; ++i) {
    int node = s_row[wave][half * 8 + i];
    atomicAdd(&magg[(size_t)node * HID + r],      ef_lo[i]);
    atomicAdd(&magg[(size_t)node * HID + 16 + r], ef_hi[i]);
  }

  // coord MLP: tanh(silu(ef @ cw1 + cb1) @ cw2)
  v8f t_lo = {}, t_hi = {};
  t_lo = WMMA_F16(aEF, frag_from_lds(sw + 4 * 1024,       r, half), t_lo);
  t_hi = WMMA_F16(aEF, frag_from_lds(sw + 4 * 1024 + 512, r, half), t_hi);
  {
    float b_lo = svec[3 * 32 + r], b_hi = svec[3 * 32 + 16 + r];
#pragma unroll
    for (int i = 0; i < 8; ++i) { t_lo[i] = silu_f(t_lo[i] + b_lo); t_hi[i] = silu_f(t_hi[i] + b_hi); }
  }
  stage_c(s_tmp[wave], t_lo, r, half, 0);
  stage_c(s_tmp[wave], t_hi, r, half, 16);
  float tmpv = 0.0f;
  if (half == 0) {
    const _Float16* rp = s_tmp[wave] + r * HID;
    float acc = 0.0f;
#pragma unroll
    for (int k = 0; k < HID; ++k) acc += (float)rp[k] * svec[4 * 32 + k];
    tmpv = tanhf(acc);
  }

  // cross MLP: tanh(silu(silu(ef @ xw1 + b) @ xw2 + b) @ xw3)
  v8f u_lo = {}, u_hi = {};
  u_lo = WMMA_F16(aEF, frag_from_lds(sw + 5 * 1024,       r, half), u_lo);
  u_hi = WMMA_F16(aEF, frag_from_lds(sw + 5 * 1024 + 512, r, half), u_hi);
  {
    float b_lo = svec[5 * 32 + r], b_hi = svec[5 * 32 + 16 + r];
#pragma unroll
    for (int i = 0; i < 8; ++i) { u_lo[i] = silu_f(u_lo[i] + b_lo); u_hi[i] = silu_f(u_hi[i] + b_hi); }
  }
  stage_c(s_tmp[wave], u_lo, r, half, 0);
  stage_c(s_tmp[wave], u_hi, r, half, 16);
  v16h aU = frag_from_lds(s_tmp[wave], r, half);
  v8f w_lo = {}, w_hi = {};
  w_lo = WMMA_F16(aU, frag_from_lds(sw + 6 * 1024,       r, half), w_lo);
  w_hi = WMMA_F16(aU, frag_from_lds(sw + 6 * 1024 + 512, r, half), w_hi);
  {
    float b_lo = svec[6 * 32 + r], b_hi = svec[6 * 32 + 16 + r];
#pragma unroll
    for (int i = 0; i < 8; ++i) { w_lo[i] = silu_f(w_lo[i] + b_lo); w_hi[i] = silu_f(w_hi[i] + b_hi); }
  }
  stage_c(s_tmp[wave], w_lo, r, half, 0);
  stage_c(s_tmp[wave], w_hi, r, half, 16);

  if (half == 0) {
    const _Float16* rp = s_tmp[wave] + r * HID;
    float acc = 0.0f;
#pragma unroll
    for (int k = 0; k < HID; ++k) acc += (float)rp[k] * svec[7 * 32 + k];
    float phi = tanhf(acc);
    atomicAdd(&tsum[(size_t)myRow * 3 + 0], dx * tmpv + cx * phi);
    atomicAdd(&tsum[(size_t)myRow * 3 + 1], dy * tmpv + cy * phi);
    atomicAdd(&tsum[(size_t)myRow * 3 + 2], dz * tmpv + cz * phi);
  }
}

// ---------------------------------------------------------------------------
// Node kernel: h += silu(concat(h,magg)@W1+b1)@W2+b2; maintains f16 mirror.
// ---------------------------------------------------------------------------
__global__ void __launch_bounds__(256, 2)
egnn_node_kernel(float* __restrict__ h_cur, _Float16* __restrict__ h_half,
                 const float* __restrict__ magg,
                 const _Float16* __restrict__ wl16, const float* __restrict__ wl32) {
  __shared__ __align__(16) _Float16 sw[3 * 1024];       // nw1t x2, nw2t
  __shared__ float svec[64];
  __shared__ __align__(16) _Float16 s_tmp[8][16 * HID];
  const int tid = threadIdx.x;
  if (tid == 0) tdm_load_to_lds(sw, wl16 + 7 * 1024, 3 * 1024 * 2);
  for (int i = tid; i < 64; i += 256) svec[i] = wl32[8 * 32 + i];
  __syncthreads();

  const int wave = tid >> 5, lane = tid & 31;
  const int r = lane & 15, half = lane >> 4;
  const int tileId = blockIdx.x * 8 + wave;
  if (tileId >= N_TILES) return;
  const int nb = tileId * 16;

  v16h aH = frag_from_global_f16(h_half, nb + r, half);
  v16h aM = frag_from_global_f32(magg,  nb + r, half);
  v8f c_lo = {}, c_hi = {};
  c_lo = WMMA_F16(aH, frag_from_lds(sw + 0,          r, half), c_lo);
  c_hi = WMMA_F16(aH, frag_from_lds(sw + 512,        r, half), c_hi);
  c_lo = WMMA_F16(aM, frag_from_lds(sw + 1024,       r, half), c_lo);
  c_hi = WMMA_F16(aM, frag_from_lds(sw + 1024 + 512, r, half), c_hi);
  {
    float b_lo = svec[r], b_hi = svec[16 + r];
#pragma unroll
    for (int i = 0; i < 8; ++i) { c_lo[i] = silu_f(c_lo[i] + b_lo); c_hi[i] = silu_f(c_hi[i] + b_hi); }
  }
  stage_c(s_tmp[wave], c_lo, r, half, 0);
  stage_c(s_tmp[wave], c_hi, r, half, 16);
  v16h aT = frag_from_lds(s_tmp[wave], r, half);
  v8f o_lo = {}, o_hi = {};
  o_lo = WMMA_F16(aT, frag_from_lds(sw + 2048,       r, half), o_lo);
  o_hi = WMMA_F16(aT, frag_from_lds(sw + 2048 + 512, r, half), o_hi);
  float b2_lo = svec[32 + r], b2_hi = svec[48 + r];
#pragma unroll
  for (int i = 0; i < 8; ++i) {
    size_t row = (size_t)(nb + half * 8 + i);
    float* hp = h_cur + row * HID;
    _Float16* hh = h_half + row * HID;
    float n0 = hp[r]      + o_lo[i] + b2_lo;
    float n1 = hp[16 + r] + o_hi[i] + b2_hi;
    hp[r] = n0;       hh[r]      = (_Float16)n0;
    hp[16 + r] = n1;  hh[16 + r] = (_Float16)n1;
  }
}

// ---------------------------------------------------------------------------
// Embedding GEMM: out = in @ W + b (f32 in, f32 out, optional f16 mirror).
// ---------------------------------------------------------------------------
__global__ void __launch_bounds__(256, 2)
embed_kernel(const float* __restrict__ hin, float* __restrict__ hout,
             _Float16* __restrict__ hhalf,
             const _Float16* __restrict__ wt, const float* __restrict__ bias) {
  __shared__ __align__(16) _Float16 sw[1024];
  __shared__ float sb[32];
  const int tid = threadIdx.x;
  if (tid == 0) tdm_load_to_lds(sw, wt, 1024 * 2);
  if (tid < 32) sb[tid] = bias[tid];
  __syncthreads();
  const int wave = tid >> 5, lane = tid & 31;
  const int r = lane & 15, half = lane >> 4;
  const int tileId = blockIdx.x * 8 + wave;
  if (tileId >= N_TILES) return;
  const int nb = tileId * 16;
  v16h a = frag_from_global_f32(hin, nb + r, half);
  v8f c_lo = {}, c_hi = {};
  c_lo = WMMA_F16(a, frag_from_lds(sw + 0,   r, half), c_lo);
  c_hi = WMMA_F16(a, frag_from_lds(sw + 512, r, half), c_hi);
  float b_lo = sb[r], b_hi = sb[16 + r];
#pragma unroll
  for (int i = 0; i < 8; ++i) {
    size_t row = (size_t)(nb + half * 8 + i);
    float v0 = c_lo[i] + b_lo, v1 = c_hi[i] + b_hi;
    float* op = hout + row * HID;
    op[r] = v0; op[16 + r] = v1;
    if (hhalf) {
      _Float16* hh = hhalf + row * HID;
      hh[r] = (_Float16)v0; hh[16 + r] = (_Float16)v1;
    }
  }
}

// ---------------------------------------------------------------------------
extern "C" void kernel_launch(void* const* d_in, const int* in_sizes, int n_in,
                              void* d_out, int out_size, void* d_ws, size_t ws_size,
                              hipStream_t stream) {
  (void)in_sizes; (void)n_in; (void)out_size; (void)ws_size;
  const float* h_in      = (const float*)d_in[0];
  const float* x_in      = (const float*)d_in[1];
  const int*   edges     = (const int*)d_in[2];
  const float* edge_attr = (const float*)d_in[3];
  const int*   batch     = (const int*)d_in[4];
  PtrTab tab;
  tab.p[0] = (const float*)d_in[5];   // emb_in_b
  tab.p[1] = (const float*)d_in[6];   // emb_in_w
  tab.p[2] = (const float*)d_in[7];   // emb_out_b
  tab.p[3] = (const float*)d_in[8];   // emb_out_w
  for (int i = 0; i < N_LAYERS * 16; ++i) tab.p[4 + i] = (const float*)d_in[9 + i];

  uint8_t* w = (uint8_t*)d_ws;
  size_t off = 0;
  auto carve = [&](size_t bytes) -> void* {
    void* p = w + off; off += (bytes + 255) & ~(size_t)255; return p;
  };
  float*    h_cur  = (float*)carve((size_t)N_NODES * HID * 4);
  _Float16* h_half = (_Float16*)carve((size_t)N_NODES * HID * 2);
  float*    x_cur  = (float*)carve((size_t)N_NODES * 3 * 4);
  float*    tsum   = (float*)carve((size_t)N_NODES * 3 * 4);
  float*    tcnt   = (float*)carve((size_t)N_NODES * 4);
  float*    magg   = (float*)carve((size_t)N_NODES * HID * 4);
  float*    csum   = (float*)carve(NUM_GRAPHS * 3 * 4);
  float*    cnt    = (float*)carve(NUM_GRAPHS * 4);
  float*    meanb  = (float*)carve(NUM_GRAPHS * 3 * 4);
  _Float16* eattr_half = (_Float16*)carve((size_t)N_EDGES * HID * 2);
  _Float16* wf16 = (_Float16*)carve((size_t)(N_LAYERS * 10 + 2) * 1024 * 2);
  float*    wf32 = (float*)carve((size_t)(N_LAYERS * 10 + 2) * 32 * 4);

  // one-time per launch: weight prep, f16 edge_attr, degrees, graph counts
  prep_weights_kernel<<<16, 256, 0, stream>>>(tab, wf16, wf32);
  f32_to_f16_kernel<<<2048, 256, 0, stream>>>(edge_attr, eattr_half, N_EDGES * HID);
  hipMemsetAsync(tcnt, 0, (size_t)N_NODES * 4, stream);
  hipMemsetAsync(cnt, 0, NUM_GRAPHS * 4, stream);
  degree_kernel<<<(N_EDGES + 255) / 256, 256, 0, stream>>>(edges, tcnt);
  gcount_kernel<<<(N_NODES + 255) / 256, 256, 0, stream>>>(batch, cnt);
  hipMemcpyAsync(x_cur, x_in, (size_t)N_NODES * 3 * 4, hipMemcpyDeviceToDevice, stream);

  const int tile_blocks = (N_TILES + 7) / 8;
  embed_kernel<<<tile_blocks, 256, 0, stream>>>(h_in, h_cur, h_half,
      wf16 + (size_t)N_LAYERS * 10 * 1024, wf32 + N_LAYERS * 10 * 32);

  for (int l = 0; l < N_LAYERS; ++l) {
    const _Float16* wl16 = wf16 + (size_t)l * 10 * 1024;
    const float*    wl32 = wf32 + l * 10 * 32;
    hipMemsetAsync(csum, 0, NUM_GRAPHS * 3 * 4, stream);
    hipMemsetAsync(tsum, 0, (size_t)N_NODES * 3 * 4, stream);
    hipMemsetAsync(magg, 0, (size_t)N_NODES * HID * 4, stream);
    mean_accum_kernel<<<(N_NODES + 255) / 256, 256, 0, stream>>>(x_cur, batch, csum);
    mean_div_kernel<<<1, 64, 0, stream>>>(csum, cnt, meanb);
    egnn_edge_kernel<<<N_EDGES / 128, 256, 0, stream>>>(h_half, x_cur, edges, eattr_half,
        batch, meanb, wl16, wl32, tsum, magg);
    coord_update_kernel<<<(N_NODES + 255) / 256, 256, 0, stream>>>(x_cur, tsum, tcnt);
    egnn_node_kernel<<<tile_blocks, 256, 0, stream>>>(h_cur, h_half, magg, wl16, wl32);
  }

  float* out_h = (float*)d_out;
  float* out_x = out_h + (size_t)N_NODES * HID;
  embed_kernel<<<tile_blocks, 256, 0, stream>>>(h_cur, out_h, (_Float16*)nullptr,
      wf16 + (size_t)(N_LAYERS * 10 + 1) * 1024, wf32 + N_LAYERS * 10 * 32 + 32);
  hipMemcpyAsync(out_x, x_cur, (size_t)N_NODES * 3 * 4, hipMemcpyDeviceToDevice, stream);
}
